// TransformerBlock_69630009803058
// MI455X (gfx1250) — matrix operand order repaired and checked by reading, not compiled, not run on hardware
//
#include <hip/hip_runtime.h>
#include <hip/hip_bf16.h>
#include <cstdint>

#define DEV __device__ __forceinline__

typedef __attribute__((ext_vector_type(16))) __bf16 bf16x16;
typedef __attribute__((ext_vector_type(8)))  __bf16 bf16x8;
typedef __attribute__((ext_vector_type(8)))  float  f32x8;
typedef __attribute__((ext_vector_type(4)))  int    v4i;

static constexpr int B_ = 8, T_ = 1024, D_ = 1024, H_ = 16, HD_ = 64, FF_ = 4096;
static constexpr int ROWS = B_ * T_;                 // 8192 token rows
static constexpr float EPS_ = 1e-5f;

// ---------------------------------------------------------------------------
// Optional CDNA5 async global->LDS DMA path (GLOBAL_LOAD_ASYNC_TO_LDS_B128,
// tracked by ASYNCcnt). Signature discovered from compiler diagnostics:
//   param0 = v4i* in global (__device__/addrspace(1)) space (source),
//   param1 = v4i* in LDS (__shared__/addrspace(3)) space (dest),
//   param2 = imm offset, param3 = imm cpol.
// Guarded so compilation falls back to the load+ds_store path.
// ---------------------------------------------------------------------------
#if defined(__has_builtin)
#if __has_builtin(__builtin_amdgcn_global_load_async_to_lds_b128) && \
    __has_builtin(__builtin_amdgcn_s_wait_asynccnt)
#define USE_ASYNC_LDS 1
#endif
#endif
#ifndef USE_ASYNC_LDS
#define USE_ASYNC_LDS 0
#endif

#if USE_ASYNC_LDS
typedef __attribute__((address_space(1))) v4i g_v4i;
typedef __attribute__((address_space(3))) v4i l_v4i;
DEV g_v4i* to_global(const void* p) { return (g_v4i*)(uintptr_t)p; }
DEV l_v4i* to_lds(void* p)          { return (l_v4i*)(uint32_t)(uintptr_t)p; }
#endif

// ---------------------------------------------------------------------------
// WMMA helpers (CDNA5 wave32, V_WMMA_F32_16X16X32_BF16)
// ---------------------------------------------------------------------------
DEV f32x8 wmma_bf16(bf16x16 a, bf16x16 b, f32x8 c) {
  // 8 args: (neg_a, A, neg_b, B, c_mod, C, reuse_a, reuse_b)
  return __builtin_amdgcn_wmma_f32_16x16x32_bf16(false, a, false, b, (short)0, c,
                                                 false, false);
}

// A-matrix fragment (16x32 bf16). Per ISA 7.12.2: lane&15 = M row;
// lane>>4 selects K-half: elems 0..7 = K(half*8..+7), 8..15 = K(16+half*8..).
DEV bf16x16 frag_a(const __bf16* rowbase, int hlf) {
  bf16x8 lo = *(const bf16x8*)(rowbase + hlf * 8);
  bf16x8 hi = *(const bf16x8*)(rowbase + 16 + hlf * 8);
  bf16x16 r;
#pragma unroll
  for (int i = 0; i < 8; ++i) { r[i] = lo[i]; r[i + 8] = hi[i]; }
  return r;
}

// B-matrix fragment (32x16 bf16). lane&15 = N column; lane>>4 selects K-half:
// elems 0..7 = K(half*8..+7), 8..15 = K(16+half*8..) (strided LDS reads).
DEV bf16x16 frag_b(const __bf16* p, int ld, int hlf) {
  bf16x16 r;
#pragma unroll
  for (int i = 0; i < 8; ++i) { r[i] = p[(hlf * 8 + i) * ld]; r[i + 8] = p[(16 + hlf * 8 + i) * ld]; }
  return r;
}

// ---------------------------------------------------------------------------
// fp32 -> bf16 weight cast (weights stay [K][N] row-major == B operand layout)
// ---------------------------------------------------------------------------
__global__ __launch_bounds__(256) void cast_bf16_kernel(const float* __restrict__ in,
                                                        __bf16* __restrict__ out, int n) {
  const int i = (blockIdx.x * 256 + threadIdx.x) * 4;
  if (i + 3 < n) {
    const float4 f = *(const float4*)(in + i);
    out[i]     = (__bf16)f.x;
    out[i + 1] = (__bf16)f.y;
    out[i + 2] = (__bf16)f.z;
    out[i + 3] = (__bf16)f.w;
  }
}

// ---------------------------------------------------------------------------
// LayerNorm (unbiased variance, ddof=1) -> fp32 (residual base) + bf16 (GEMM A)
// ---------------------------------------------------------------------------
__global__ __launch_bounds__(256) void ln_kernel(const float* __restrict__ x,
                                                 const float* __restrict__ g,
                                                 const float* __restrict__ be,
                                                 float* __restrict__ y,
                                                 __bf16* __restrict__ yb) {
  __shared__ float sred[256];
  const int row = blockIdx.x, tid = threadIdx.x;
  const float* xr = x + (size_t)row * D_;
  float v[4];
  float s = 0.f;
#pragma unroll
  for (int j = 0; j < 4; ++j) { v[j] = xr[tid + j * 256]; s += v[j]; }
  sred[tid] = s; __syncthreads();
  for (int o = 128; o > 0; o >>= 1) { if (tid < o) sred[tid] += sred[tid + o]; __syncthreads(); }
  const float mean = sred[0] * (1.0f / D_);
  __syncthreads();
  float ss = 0.f;
#pragma unroll
  for (int j = 0; j < 4; ++j) { const float d = v[j] - mean; ss += d * d; }
  sred[tid] = ss; __syncthreads();
  for (int o = 128; o > 0; o >>= 1) { if (tid < o) sred[tid] += sred[tid + o]; __syncthreads(); }
  const float rstd = rsqrtf(sred[0] * (1.0f / (D_ - 1)) + EPS_);
#pragma unroll
  for (int j = 0; j < 4; ++j) {
    const int i = tid + j * 256;
    const float o = (v[j] - mean) * rstd * g[i] + be[i];
    y[(size_t)row * D_ + i]  = o;
    yb[(size_t)row * D_ + i] = (__bf16)o;
  }
}

// ---------------------------------------------------------------------------
// Generic bf16 WMMA GEMM: C[M][N] = A[M][K] * B[K][N] (+bias, +GELU, +residual)
// 256 threads = 8 waves; tile 128x128x32; wave -> 32x64 (2x4 WMMA tiles).
// Staging uses async global->LDS DMA when available.
// ---------------------------------------------------------------------------
template <bool GELU, bool RES, bool OUTBF>
__global__ __launch_bounds__(256) void gemm_bf16_kernel(
    const __bf16* __restrict__ A, const __bf16* __restrict__ Bw,
    const float* __restrict__ bias, const float* __restrict__ res,
    float* __restrict__ Cf, __bf16* __restrict__ Cb, int M, int N, int K) {
  constexpr int BM = 128, BN = 128, BK = 32;
  __shared__ __bf16 As[BM][BK];
  __shared__ __bf16 Bs[BK][BN];
  const int tid = threadIdx.x;
  const int lane = tid & 31, wv = tid >> 5;
  const int hlf = lane >> 4, r16 = lane & 15;
  const int m0 = blockIdx.y * BM, n0 = blockIdx.x * BN;
  const int wm = wv & 3, wn = wv >> 2;  // 4 waves along M, 2 along N

  f32x8 acc[2][4];
#pragma unroll
  for (int i = 0; i < 2; ++i)
#pragma unroll
    for (int j = 0; j < 4; ++j) acc[i][j] = 0.0f;

  const int ar = tid >> 1, ac = (tid & 1) * 16;  // A stage: 128 rows x 2x16
  const int br = tid >> 3, bc = (tid & 7) * 16;  // B stage: 32 rows x 8x16

  for (int k0 = 0; k0 < K; k0 += BK) {
    __syncthreads();
#if USE_ASYNC_LDS
    __builtin_amdgcn_global_load_async_to_lds_b128(
        to_global(&A[(size_t)(m0 + ar) * K + k0 + ac]), to_lds(&As[ar][ac]), 0, 0);
    __builtin_amdgcn_global_load_async_to_lds_b128(
        to_global(&A[(size_t)(m0 + ar) * K + k0 + ac + 8]), to_lds(&As[ar][ac + 8]), 0, 0);
    __builtin_amdgcn_global_load_async_to_lds_b128(
        to_global(&Bw[(size_t)(k0 + br) * N + n0 + bc]), to_lds(&Bs[br][bc]), 0, 0);
    __builtin_amdgcn_global_load_async_to_lds_b128(
        to_global(&Bw[(size_t)(k0 + br) * N + n0 + bc + 8]), to_lds(&Bs[br][bc + 8]), 0, 0);
    __builtin_amdgcn_s_wait_asynccnt(0);
#else
    *(bf16x8*)&As[ar][ac]     = *(const bf16x8*)&A[(size_t)(m0 + ar) * K + k0 + ac];
    *(bf16x8*)&As[ar][ac + 8] = *(const bf16x8*)&A[(size_t)(m0 + ar) * K + k0 + ac + 8];
    *(bf16x8*)&Bs[br][bc]     = *(const bf16x8*)&Bw[(size_t)(k0 + br) * N + n0 + bc];
    *(bf16x8*)&Bs[br][bc + 8] = *(const bf16x8*)&Bw[(size_t)(k0 + br) * N + n0 + bc + 8];
#endif
    if (k0 + BK < K) {
      __builtin_prefetch(&A[(size_t)(m0 + ar) * K + k0 + BK + ac], 0, 0);
      __builtin_prefetch(&Bw[(size_t)(k0 + BK + br) * N + n0 + bc], 0, 0);
    }
    __syncthreads();

    bf16x16 af[2], bfv[4];
#pragma unroll
    for (int mi = 0; mi < 2; ++mi)
      af[mi] = frag_a(&As[wm * 32 + mi * 16 + r16][0], hlf);
#pragma unroll
    for (int ni = 0; ni < 4; ++ni)
      bfv[ni] = frag_b(&Bs[0][wn * 64 + ni * 16 + r16], BN, hlf);
#pragma unroll
    for (int mi = 0; mi < 2; ++mi)
#pragma unroll
      for (int ni = 0; ni < 4; ++ni)
        acc[mi][ni] = wmma_bf16(af[mi], bfv[ni], acc[mi][ni]);
  }

  // Epilogue: C layout -> VGPR j holds M = j + 8*(lane>>4); N = lane&15.
#pragma unroll
  for (int mi = 0; mi < 2; ++mi) {
#pragma unroll
    for (int ni = 0; ni < 4; ++ni) {
      const int mg0 = m0 + wm * 32 + mi * 16 + hlf * 8;
      const int ng  = n0 + wn * 64 + ni * 16 + r16;
      const float bb = bias ? bias[ng] : 0.0f;
#pragma unroll
      for (int j = 0; j < 8; ++j) {
        const size_t idx = (size_t)(mg0 + j) * N + ng;
        float vv = acc[mi][ni][j] + bb;
        if (GELU) vv = 0.5f * vv * (1.0f + erff(vv * 0.70710678118654752f));
        if (RES)  vv += res[idx];
        if (OUTBF) Cb[idx] = (__bf16)vv; else Cf[idx] = vv;
      }
    }
  }
}

// ---------------------------------------------------------------------------
// Flash-style attention: block = (b, h, 32 query rows). Scores live entirely
// in LDS (32x1024 fp32 = 128 KB; CDNA5 WGP has 320 KB). Both GEMMs use WMMA.
// ---------------------------------------------------------------------------
static constexpr int SMEM_S   = 32 * 1024 * 4;   // fp32 scores
static constexpr int SMEM_QS  = 32 * 64 * 2;     // Q tile / P block (aliased)
static constexpr int SMEM_KS  = 64 * 64 * 2;     // K^T tile / V tile (aliased)
static constexpr int SMEM_RED = 32 * 8 * 4;      // row reductions
static constexpr int ATT_SMEM = SMEM_S + SMEM_QS + SMEM_KS + SMEM_RED + 32 * 4 + 32 * 4;

__global__ __launch_bounds__(256) void attn_kernel(const __bf16* __restrict__ q,
                                                   const __bf16* __restrict__ k,
                                                   const __bf16* __restrict__ v,
                                                   __bf16* __restrict__ ctx) {
  extern __shared__ char smem[];
  float*  S   = (float*)smem;
  __bf16* Qs  = (__bf16*)(smem + SMEM_S);
  __bf16* Pb  = Qs;  // alias: Q dead after phase 1
  __bf16* Ks  = (__bf16*)(smem + SMEM_S + SMEM_QS);
  __bf16* Vs  = Ks;  // alias: K dead after phase 1
  float*  red = (float*)(smem + SMEM_S + SMEM_QS + SMEM_KS);
  float*  rm  = red + 32 * 8;
  float*  rs  = rm + 32;

  const int tid = threadIdx.x, lane = tid & 31, wv = tid >> 5;
  const int hlf = lane >> 4, r16 = lane & 15;
  const int mi = wv & 1, ni = wv >> 1;       // 8 waves cover 2x4 16x16 tiles
  const int bh = blockIdx.y, b = bh >> 4, h = bh & 15;
  const int r0 = blockIdx.x * 32;
  const size_t base = ((size_t)b * T_) * D_ + (size_t)h * HD_;

  // Stage Q (32 x 64, row-major == A layout)
  {
    const int tl = tid >> 3, c0 = (tid & 7) * 8;
    *(bf16x8*)&Qs[tl * 64 + c0] =
        *(const bf16x8*)&q[base + (size_t)(r0 + tl) * D_ + c0];
  }

  // Phase 1: S = (Q * K^T) * HD^-0.5 ; K staged transposed to [hd][t].
  for (int kb = 0; kb < 16; ++kb) {
    __syncthreads();
    {
      const int tl = tid >> 2, c0 = (tid & 3) * 16;
      const __bf16* kr = &k[base + (size_t)(kb * 64 + tl) * D_];
#pragma unroll
      for (int j = 0; j < 16; ++j) Ks[(c0 + j) * 64 + tl] = kr[c0 + j];
    }
    __syncthreads();
    f32x8 a0 = 0.0f;
#pragma unroll
    for (int kk = 0; kk < 2; ++kk) {
      bf16x16 af  = frag_a(&Qs[(mi * 16 + r16) * 64 + kk * 32], hlf);
      bf16x16 bf_ = frag_b(&Ks[(kk * 32) * 64 + ni * 16 + r16], 64, hlf);
      a0 = wmma_bf16(af, bf_, a0);
    }
    const int col = kb * 64 + ni * 16 + r16;
    const int rb  = mi * 16 + hlf * 8;
#pragma unroll
    for (int j = 0; j < 8; ++j) S[(rb + j) * 1024 + col] = a0[j] * 0.125f;
  }
  __syncthreads();

  // Phase 2: softmax per row (8 threads per row).
  {
    const int sr = tid >> 3, part = tid & 7;
    float mx = -3.0e38f;
    for (int c = part * 128; c < part * 128 + 128; ++c)
      mx = fmaxf(mx, S[sr * 1024 + c]);
    red[sr * 8 + part] = mx; __syncthreads();
    if (part == 0) {
      float m2 = red[sr * 8];
#pragma unroll
      for (int i = 1; i < 8; ++i) m2 = fmaxf(m2, red[sr * 8 + i]);
      rm[sr] = m2;
    }
    __syncthreads();
    const float m = rm[sr];
    float sum = 0.f;
    for (int c = part * 128; c < part * 128 + 128; ++c) {
      const float e = __expf(S[sr * 1024 + c] - m);
      S[sr * 1024 + c] = e;
      sum += e;
    }
    red[sr * 8 + part] = sum; __syncthreads();
    if (part == 0) {
      float t2 = red[sr * 8];
#pragma unroll
      for (int i = 1; i < 8; ++i) t2 += red[sr * 8 + i];
      rs[sr] = 1.0f / t2;
    }
  }

  // Phase 3: ctx = P * V ; P re-quantized to bf16 per 64-col block.
  f32x8 a1 = 0.0f;
  for (int kb = 0; kb < 16; ++kb) {
    __syncthreads();
    {
      const int tl = tid >> 3, c0 = (tid & 7) * 8;
      const float inv = rs[tl];
#pragma unroll
      for (int j = 0; j < 8; ++j)
        Pb[tl * 64 + c0 + j] = (__bf16)(S[tl * 1024 + kb * 64 + c0 + j] * inv);
    }
    {
      const int tl = tid >> 2, c0 = (tid & 3) * 16;
      const __bf16* vr = &v[base + (size_t)(kb * 64 + tl) * D_];
      *(bf16x8*)&Vs[tl * 64 + c0]     = *(const bf16x8*)&vr[c0];
      *(bf16x8*)&Vs[tl * 64 + c0 + 8] = *(const bf16x8*)&vr[c0 + 8];
    }
    __syncthreads();
#pragma unroll
    for (int kk = 0; kk < 2; ++kk) {
      bf16x16 af  = frag_a(&Pb[(mi * 16 + r16) * 64 + kk * 32], hlf);
      bf16x16 bf_ = frag_b(&Vs[(kk * 32) * 64 + ni * 16 + r16], 64, hlf);
      a1 = wmma_bf16(af, bf_, a1);
    }
  }
  const int t0 = r0 + mi * 16 + hlf * 8;
  const int hd = ni * 16 + r16;
#pragma unroll
  for (int j = 0; j < 8; ++j)
    ctx[base + (size_t)(t0 + j) * D_ + hd] = (__bf16)a1[j];
}

// ---------------------------------------------------------------------------
// Host orchestration
// ---------------------------------------------------------------------------
extern "C" void kernel_launch(void* const* d_in, const int* in_sizes, int n_in,
                              void* d_out, int out_size, void* d_ws, size_t ws_size,
                              hipStream_t stream) {
  (void)in_sizes; (void)n_in; (void)out_size; (void)ws_size;
  const float* x   = (const float*)d_in[0];
  const float* Wq  = (const float*)d_in[1];  const float* bq  = (const float*)d_in[2];
  const float* Wk  = (const float*)d_in[3];  const float* bk  = (const float*)d_in[4];
  const float* Wv  = (const float*)d_in[5];  const float* bv  = (const float*)d_in[6];
  const float* Wo  = (const float*)d_in[7];  const float* bo  = (const float*)d_in[8];
  const float* W1  = (const float*)d_in[9];  const float* b1  = (const float*)d_in[10];
  const float* W2  = (const float*)d_in[11]; const float* b2  = (const float*)d_in[12];
  const float* g1  = (const float*)d_in[13]; const float* be1 = (const float*)d_in[14];
  const float* g2  = (const float*)d_in[15]; const float* be2 = (const float*)d_in[16];
  float* out = (float*)d_out;

  char* ws = (char*)d_ws;
  size_t off = 0;
  auto carve = [&](size_t bytes) -> void* {
    void* p = ws + off;
    off += (bytes + 255) & ~(size_t)255;
    return p;
  };
  float*  h    = (float*) carve((size_t)ROWS * D_ * 4);
  __bf16* hb   = (__bf16*)carve((size_t)ROWS * D_ * 2);
  float*  h2   = (float*) carve((size_t)ROWS * D_ * 4);
  float*  h2n  = (float*) carve((size_t)ROWS * D_ * 4);
  __bf16* h2nb = (__bf16*)carve((size_t)ROWS * D_ * 2);
  __bf16* qb   = (__bf16*)carve((size_t)ROWS * D_ * 2);
  __bf16* kb_  = (__bf16*)carve((size_t)ROWS * D_ * 2);
  __bf16* vb   = (__bf16*)carve((size_t)ROWS * D_ * 2);
  __bf16* ctxb = (__bf16*)carve((size_t)ROWS * D_ * 2);
  __bf16* ub   = (__bf16*)carve((size_t)ROWS * FF_ * 2);
  __bf16* Wqb  = (__bf16*)carve((size_t)D_ * D_ * 2);
  __bf16* Wkb  = (__bf16*)carve((size_t)D_ * D_ * 2);
  __bf16* Wvb  = (__bf16*)carve((size_t)D_ * D_ * 2);
  __bf16* Wob  = (__bf16*)carve((size_t)D_ * D_ * 2);
  __bf16* W1b  = (__bf16*)carve((size_t)D_ * FF_ * 2);
  __bf16* W2b  = (__bf16*)carve((size_t)FF_ * D_ * 2);

  const dim3 blk(256);

  // Weight casts (bf16, [K][N] layout == WMMA B operand staging layout)
  cast_bf16_kernel<<<dim3(D_ * D_ / 1024),  blk, 0, stream>>>(Wq, Wqb, D_ * D_);
  cast_bf16_kernel<<<dim3(D_ * D_ / 1024),  blk, 0, stream>>>(Wk, Wkb, D_ * D_);
  cast_bf16_kernel<<<dim3(D_ * D_ / 1024),  blk, 0, stream>>>(Wv, Wvb, D_ * D_);
  cast_bf16_kernel<<<dim3(D_ * D_ / 1024),  blk, 0, stream>>>(Wo, Wob, D_ * D_);
  cast_bf16_kernel<<<dim3(D_ * FF_ / 1024), blk, 0, stream>>>(W1, W1b, D_ * FF_);
  cast_bf16_kernel<<<dim3(FF_ * D_ / 1024), blk, 0, stream>>>(W2, W2b, FF_ * D_);

  // LN1
  ln_kernel<<<dim3(ROWS), blk, 0, stream>>>(x, g1, be1, h, hb);

  // QKV projections (bf16 out)
  const dim3 gdd(D_ / 128, ROWS / 128);
  gemm_bf16_kernel<false, false, true><<<gdd, blk, 0, stream>>>(
      hb, Wqb, bq, nullptr, nullptr, qb, ROWS, D_, D_);
  gemm_bf16_kernel<false, false, true><<<gdd, blk, 0, stream>>>(
      hb, Wkb, bk, nullptr, nullptr, kb_, ROWS, D_, D_);
  gemm_bf16_kernel<false, false, true><<<gdd, blk, 0, stream>>>(
      hb, Wvb, bv, nullptr, nullptr, vb, ROWS, D_, D_);

  // Attention (LDS-resident scores; ~141 KB dynamic LDS per workgroup)
  (void)hipFuncSetAttribute((const void*)attn_kernel,
                            hipFuncAttributeMaxDynamicSharedMemorySize, ATT_SMEM);
  attn_kernel<<<dim3(T_ / 32, B_ * H_), blk, ATT_SMEM, stream>>>(qb, kb_, vb, ctxb);

  // Output projection + residual (h2 = h + ctx@Wo + bo), fp32 out
  gemm_bf16_kernel<false, true, false><<<gdd, blk, 0, stream>>>(
      ctxb, Wob, bo, h, h2, nullptr, ROWS, D_, D_);

  // LN2
  ln_kernel<<<dim3(ROWS), blk, 0, stream>>>(h2, g2, be2, h2n, h2nb);

  // FFN: u = GELU(h2n@W1 + b1) bf16 ; out = h2n + u@W2 + b2 fp32
  gemm_bf16_kernel<true, false, true><<<dim3(FF_ / 128, ROWS / 128), blk, 0, stream>>>(
      h2nb, W1b, b1, nullptr, nullptr, ub, ROWS, FF_, D_);
  gemm_bf16_kernel<false, true, false><<<gdd, blk, 0, stream>>>(
      ub, W2b, b2, h2n, out, nullptr, ROWS, D_, FF_);
}
